// EdgeNetwork_30425548325107
// MI455X (gfx1250) — compile-verified, matrix operand order checked by reading, not run on hardware
//
#include <hip/hip_runtime.h>
#include <hip/hip_bf16.h>
#include <math.h>

#define B_      2
#define N_      4096
#define E_      8192
#define F_      16
#define H_      64
#define NCHUNK  8
#define NPER    (N_ / NCHUNK)   // 512 rows per chunk

typedef __attribute__((ext_vector_type(4)))  float    f4;
typedef __attribute__((ext_vector_type(8)))  float    v8f;
typedef __attribute__((ext_vector_type(16))) _Float16 v16h;

// Branch-free tanh: 1 - 2/(1+exp(2x)).  v_exp_f32 + v_rcp_f32, no EXEC churn.
// Exact limits (exp->0 => -1, exp->inf => +1); ~1e-6 error elsewhere, well
// under the f16 WMMA input quantization already accepted.
__device__ __forceinline__ float fast_tanh(float x) {
  float ex = __expf(2.0f * x);
  return 1.0f - __fdividef(2.0f, ex + 1.0f);
}

// ---------------------------------------------------------------------------
// Phase 1: recover gather indices from the dense one-hot matrices.
// idx(b,m,e) = sum_n n * R[b,n,e]  (exact in fp32: single 1.0 per column).
// Thread owns 4 consecutive edges -> coalesced b128 non-temporal loads.
// pidx layout: [b][m][chunk][E] float partial sums (m: 0=Ro, 1=Ri).
// This phase reads 512 MB and sets the kernel's runtime floor (~22 us @ 23.3 TB/s).
// ---------------------------------------------------------------------------
__global__ __launch_bounds__(256) void recover_idx_kernel(
    const float* __restrict__ Ri, const float* __restrict__ Ro,
    float* __restrict__ pidx) {
  int tid   = blockIdx.x * 256 + threadIdx.x;   // 0 .. 65535
  int e4    = tid & (E_ / 4 - 1);               // edge-quad (2048 per matrix)
  int rest  = tid >> 11;
  int chunk = rest & (NCHUNK - 1);
  rest    >>= 3;
  int m = rest & 1;                             // 0 = Ro, 1 = Ri
  int b = rest >> 1;

  const float* R = (m == 0) ? Ro : Ri;
  const f4* base = (const f4*)(R + (size_t)b * N_ * E_) + e4;

  int n0 = chunk * NPER;
  f4 acc = {0.f, 0.f, 0.f, 0.f};
#pragma unroll 8
  for (int n = n0; n < n0 + NPER; ++n) {
    f4 v = __builtin_nontemporal_load(base + (size_t)n * (E_ / 4));
    float fn = (float)n;
    acc.x = fmaf(v.x, fn, acc.x);
    acc.y = fmaf(v.y, fn, acc.y);
    acc.z = fmaf(v.z, fn, acc.z);
    acc.w = fmaf(v.w, fn, acc.w);
  }
  f4* outp = (f4*)(pidx + ((((size_t)b * 2 + m) * NCHUNK) + chunk) * E_) + e4;
  *outp = acc;
}

// ---------------------------------------------------------------------------
// Phase 2: one wave32 per 16-edge tile.
//   feats[16 x 32] (A, f16)  @  W1[32 x 64] (4 B-tiles of 32x16)  -> h[16 x 64]
//   via v_wmma_f32_16x16x32_f16, then tanh, h @ W2 + b2, sigmoid.
// A layout (16-bit 16x32, ISA 7.12.2): lane = (M | hi<<4); halves 0..7 hold
// K = hi*8 + 0..7, halves 8..15 hold K = 16 + hi*8 + 0..7.  B mirrors with N.
// C/D: VGPR r, lanes 0-15 -> M=r, lanes 16-31 -> M=r+8, N = lane&15.
// ---------------------------------------------------------------------------
__global__ __launch_bounds__(32) void edge_mlp_kernel(
    const float* __restrict__ X, const float* __restrict__ pidx,
    const float* __restrict__ W1, const float* __restrict__ b1,
    const float* __restrict__ W2, const float* __restrict__ b2,
    float* __restrict__ out) {
  const int lane = threadIdx.x;
  const int tile = blockIdx.x;
  const int b    = tile / (E_ / 16);
  const int e0   = (tile % (E_ / 16)) * 16;

  __shared__ float s_feat[16][32];   // [edge][k]  (k<16: Ro-gather, k>=16: Ri)
  __shared__ int   s_idx[32];        // lanes 0-15: Ro idx, 16-31: Ri idx
  __shared__ float s_part[16][16];   // [edge][lane-column partial]

  const int m  = lane >> 4;          // 0 = Ro features, 1 = Ri features
  const int lo = lane & 15;

  // --- sum chunk partials -> node index (exact integer in fp32) ---
  {
    const float* p = pidx + (((size_t)b * 2 + m) * NCHUNK) * E_ + (e0 + lo);
    float s = 0.f;
#pragma unroll
    for (int c = 0; c < NCHUNK; ++c) s += p[(size_t)c * E_];
    s_idx[lane] = (int)(s + 0.5f);
  }
  __syncthreads();

  // --- gather X rows into the concatenated feature tile ---
  {
    int node = s_idx[lane];
    const float* xr = X + ((size_t)b * N_ + node) * F_;
#pragma unroll
    for (int k = 0; k < F_; ++k) s_feat[lo][m * F_ + k] = xr[k];
  }
  __syncthreads();

  // --- pack A operand (feats, f16) per documented 16-bit 16x32 layout ---
  v16h A;
#pragma unroll
  for (int v = 0; v < 4; ++v) {
    A[2 * v]         = (_Float16)s_feat[lo][m * 8 + 2 * v];
    A[2 * v + 1]     = (_Float16)s_feat[lo][m * 8 + 2 * v + 1];
    A[8 + 2 * v]     = (_Float16)s_feat[lo][16 + m * 8 + 2 * v];
    A[8 + 2 * v + 1] = (_Float16)s_feat[lo][16 + m * 8 + 2 * v + 1];
  }

  float partial[8];
#pragma unroll
  for (int r = 0; r < 8; ++r) partial[r] = 0.f;

#pragma unroll
  for (int t = 0; t < 4; ++t) {
    const int col = t * 16 + lo;     // hidden unit handled by this lane column
    v16h Bm;
#pragma unroll
    for (int v = 0; v < 4; ++v) {
      Bm[2 * v]         = (_Float16)W1[(m * 8 + 2 * v) * H_ + col];
      Bm[2 * v + 1]     = (_Float16)W1[(m * 8 + 2 * v + 1) * H_ + col];
      Bm[8 + 2 * v]     = (_Float16)W1[(16 + m * 8 + 2 * v) * H_ + col];
      Bm[8 + 2 * v + 1] = (_Float16)W1[(16 + m * 8 + 2 * v + 1) * H_ + col];
    }
    v8f C = {};
    C = __builtin_amdgcn_wmma_f32_16x16x32_f16(
        /*neg_a=*/false, A, /*neg_b=*/false, Bm,
        /*c_mod=*/(short)0, C, /*reuse_a=*/false, /*reuse_b=*/false);

    const float b1c = b1[col];
    const float w2c = W2[col];
#pragma unroll
    for (int r = 0; r < 8; ++r) {
      float h = fast_tanh(C[r] + b1c);
      partial[r] = fmaf(h, w2c, partial[r]);   // edge M = r + 8*m
    }
  }

#pragma unroll
  for (int r = 0; r < 8; ++r) s_part[r + 8 * m][lo] = partial[r];
  __syncthreads();

  if (lane < 16) {
    float s = 0.f;
#pragma unroll
    for (int j = 0; j < 16; ++j) s += s_part[lane][j];
    s += b2[0];
    out[(size_t)b * E_ + e0 + lane] = __fdividef(1.0f, 1.0f + __expf(-s));
  }
}

// ---------------------------------------------------------------------------
extern "C" void kernel_launch(void* const* d_in, const int* in_sizes, int n_in,
                              void* d_out, int out_size, void* d_ws, size_t ws_size,
                              hipStream_t stream) {
  (void)in_sizes; (void)n_in; (void)out_size; (void)ws_size;
  const float* X  = (const float*)d_in[0];
  const float* Ri = (const float*)d_in[1];
  const float* Ro = (const float*)d_in[2];
  const float* W1 = (const float*)d_in[3];
  const float* b1 = (const float*)d_in[4];
  const float* W2 = (const float*)d_in[5];
  const float* b2 = (const float*)d_in[6];
  float* out  = (float*)d_out;
  float* pidx = (float*)d_ws;   // B*2*NCHUNK*E floats = 1 MB scratch

  const int threads1 = B_ * 2 * NCHUNK * (E_ / 4);          // 65536
  recover_idx_kernel<<<threads1 / 256, 256, 0, stream>>>(Ri, Ro, pidx);
  edge_mlp_kernel<<<B_ * (E_ / 16), 32, 0, stream>>>(X, pidx, W1, b1, W2, b2, out);
}
